// RefineBlcok0_70944269796228
// MI455X (gfx1250) — matrix operand order repaired and checked by reading, not compiled, not run on hardware
//
#include <hip/hip_runtime.h>
#include <hip/hip_bf16.h>

typedef __attribute__((ext_vector_type(16))) _Float16 v16h;
typedef __attribute__((ext_vector_type(8)))  _Float16 v8h;
typedef __attribute__((ext_vector_type(8)))  float    v8f;
typedef __attribute__((ext_vector_type(4)))  unsigned int u32x4;
typedef __attribute__((ext_vector_type(8)))  int      i32x8;
typedef __attribute__((ext_vector_type(4)))  int      i32x4;

#define B_DIM 8
#define C_CH  128
#define H_DIM 128
#define W_DIM 128
#define HW    (H_DIM * W_DIM)           // 16384
#define CHW   (C_CH * HW)               // 2097152
#define NTOT  ((size_t)B_DIM * CHW)     // 16777216
#define EPSF  1e-5f
#define SLOPE 0.2f

// padded pixel-major activation tensor: [B][130*130][128] f16 (1-px halo ring)
#define PH    130
#define PHW   (PH * PH)                 // 16900
// LDS channel pitch (f16 units): 272B row stride = 16B-aligned, bank-rotating
#define CP    136

// ---------------------------------------------------------------------------
// Weight pre-pack: f32 [O][C][taps] -> f16 in per-lane WMMA A-fragment order.
// packed[(((ocgrp*taps + t)*4 + cg)*32 + lane)*16 + e]
//   lane = hi*16 + m ; K = hi*16+e
// ---------------------------------------------------------------------------
__global__ __launch_bounds__(256)
void pack_w_kernel(const float* __restrict__ w, _Float16* __restrict__ pw, int taps)
{
    int idx = blockIdx.x * 256 + threadIdx.x;
    int total = 16384 * taps;           // 8*taps*4*32*16
    if (idx >= total) return;
    int e    = idx & 15;
    int lane = (idx >> 4) & 31;
    int cg   = (idx >> 9) & 3;
    int rest = idx >> 11;
    int t    = rest % taps;
    int og   = rest / taps;
    int m  = lane & 15;
    int hi = lane >> 4;
    int k  = hi * 16 + e;
    pw[idx] = (_Float16)w[((og * 16 + m) * C_CH + cg * 32 + k) * taps + t];
}

// ---------------------------------------------------------------------------
// TDM helpers (probe-verified builtin; __has_builtin guarded, wave-level issue)
// Loads a 2D tile (dim0 contiguous f16, dim1 rows) into LDS, inserting 4
// DWORDs of padding every 64 DWORDs (-> 128 f16 data + 8 f16 pad = CP pitch).
// ---------------------------------------------------------------------------
__device__ __forceinline__
void tdm_load_2d(void* lds_dst, const _Float16* gsrc,
                 unsigned tdim0, unsigned tdim1, unsigned til0, unsigned til1,
                 unsigned stride0)
{
#if __has_builtin(__builtin_amdgcn_tensor_load_to_lds)
    unsigned long long ga = (unsigned long long)(const char*)gsrc;
    unsigned lds_off = (unsigned)(unsigned long long)lds_dst;   // low 32b = LDS addr
    u32x4 g0 = { 1u,                                            // count = 1
                 lds_off,
                 (unsigned)(ga & 0xffffffffu),
                 (unsigned)(((ga >> 32) & 0x01ffffffu) | (2u << 30)) }; // type=2
    // data_size=2B, pad_enable, pad_interval=64DW(code5), pad_amount=4DW(code3)
    i32x8 g1 = { (int)(0x10000u | (1u << 20) | (5u << 22) | (3u << 25)),
                 (int)((tdim0 & 0xffffu) << 16),
                 (int)(((tdim0 >> 16) & 0xffffu) | ((tdim1 & 0xffffu) << 16)),
                 (int)(((tdim1 >> 16) & 0xffffu) | ((til0 & 0xffffu) << 16)),
                 (int)(til1 & 0xffffu),
                 (int)stride0, 0, 0 };
    i32x4 gz = { 0, 0, 0, 0 };
#if defined(__clang_major__) && __clang_major__ >= 23
    i32x8 gz8 = { 0, 0, 0, 0, 0, 0, 0, 0 };
    __builtin_amdgcn_tensor_load_to_lds(g0, g1, gz, gz, gz8, 0);
#else
    __builtin_amdgcn_tensor_load_to_lds(g0, g1, gz, gz, 0);
#endif
#if __has_builtin(__builtin_amdgcn_s_wait_tensorcnt)
    __builtin_amdgcn_s_wait_tensorcnt(0);
#else
    asm volatile("s_wait_tensorcnt 0x0" ::: "memory");
#endif
#else
    (void)lds_dst; (void)gsrc; (void)tdim0; (void)tdim1; (void)til0; (void)til1; (void)stride0;
#endif
}

__device__ __forceinline__ v16h lds_frag(const _Float16* p)
{
    v8h lo = *(const v8h*)(p);
    v8h hi = *(const v8h*)(p + 8);
    return __builtin_shufflevector(lo, hi, 0,1,2,3,4,5,6,7,8,9,10,11,12,13,14,15);
}

// ---------------------------------------------------------------------------
// 3x3 conv, pad=1, implicit GEMM on WMMA f16.
// Block = 256 thr (8 wave32) -> (b, h, 64-px wtile), all 128 out channels.
// LDS tile pixel-major: lx[(r*66 + w)*CP + c], staged by TDM (FMT=1) or
// manual transpose (FMT=0, f32 NCHW input for conv1).
// ---------------------------------------------------------------------------
#define LDS3 (3 * 66 * CP)              // 26928 f16 = 53856 B

template <int FMT>
__global__ __launch_bounds__(256)
void conv3x3_wmma_kernel(const void* __restrict__ xin, const _Float16* __restrict__ pw,
                         const float* __restrict__ bias, float* __restrict__ y,
                         int apply_sigmoid)
{
    __shared__ _Float16 lx[LDS3];

    const int blk = blockIdx.x;         // B*H*2
    const int wt  = blk & 1;
    const int h   = (blk >> 1) & (H_DIM - 1);
    const int b   = blk >> 8;
    const int w0  = wt * 64;
    const int tid = threadIdx.x;

    __builtin_prefetch(pw + (tid << 4), 0, 3);

    if (FMT == 1) {
        // f16 padded pixel-major input: straight strided copy -> TDM
        const _Float16* src = (const _Float16*)xin +
            ((size_t)b * PHW + (size_t)h * PH + w0) * C_CH;
#if __has_builtin(__builtin_amdgcn_tensor_load_to_lds)
        if (tid < 32)
            tdm_load_2d(lx, src, /*tdim0=*/66 * C_CH, /*tdim1=*/PH,
                        /*til0=*/66 * C_CH, /*til1=*/3, /*stride0=*/PH * C_CH);
#else
        // wide manual fallback: 16B chunks, 3*66 rows of 128 f16 -> CP pitch
        for (int i = tid; i < 3 * 66 * 16; i += 256) {
            int chunk = i & 15, row = i >> 4;   // row in [0,198): r*66+w
            int r = row / 66, ww = row - r * 66;
            *(uint4*)&lx[(size_t)row * CP + chunk * 8] =
                *(const uint4*)&src[((size_t)r * PH + ww) * C_CH + chunk * 8];
        }
#endif
    } else {
        // f32 NCHW input (conv1): transpose + convert, zero halo
        const float* xb = (const float*)xin + (size_t)b * CHW;
        for (int idx = tid; idx < 3 * 66 * C_CH; idx += 256) {
            int ww   = idx % 66;
            int rest = idx / 66;
            int r    = rest % 3;
            int c    = rest / 3;
            int gh = h - 1 + r, gw = w0 - 1 + ww;
            float v = 0.f;
            if ((unsigned)gh < (unsigned)H_DIM && (unsigned)gw < (unsigned)W_DIM)
                v = xb[(size_t)c * HW + gh * W_DIM + gw];
            lx[(size_t)(r * 66 + ww) * CP + c] = (_Float16)v;
        }
    }
    __syncthreads();

    const int lane = tid & 31;
    const int wave = tid >> 5;
    const int oc0  = wave * 16;
    const int hi   = lane >> 4;
    const int n    = lane & 15;

    v8f acc[4] = {};
    const v16h* pA = (const v16h*)pw;   // packed fragments, 32B/lane coalesced

    for (int t = 0; t < 9; ++t) {
        const int ti = t / 3, tj = t % 3;
        for (int cg = 0; cg < 4; ++cg) {
            v16h a = pA[(size_t)(((wave * 9 + t) * 4 + cg) * 32 + lane)];
            const _Float16* lb0 = lx + (size_t)(ti * 66 + tj + n) * CP + cg * 32 + hi * 16;
#pragma unroll
            for (int nt = 0; nt < 4; ++nt) {
                v16h bf = lds_frag(lb0 + (size_t)nt * 16 * CP);
                acc[nt] = __builtin_amdgcn_wmma_f32_16x16x32_f16(
                    false, a, false, bf, (short)0, acc[nt], false, false);
            }
        }
    }

    float* yb = y + (size_t)b * CHW + (size_t)h * W_DIM + w0;
#pragma unroll
    for (int v = 0; v < 8; ++v) {
        const int M  = v + hi * 8;
        const float bv = bias[oc0 + M];
#pragma unroll
        for (int nt = 0; nt < 4; ++nt) {
            float o = acc[nt][v] + bv;
            if (apply_sigmoid) o = 1.f / (1.f + __expf(-o));
            yb[(size_t)(oc0 + M) * HW + nt * 16 + n] = o;
        }
    }
}

// ---------------------------------------------------------------------------
// Fused q/k/v 1x1 convs. LDS tile [64 px][CP c] staged by one TDM descriptor;
// each B-fragment shared by 3 WMMA accumulator sets.
// ---------------------------------------------------------------------------
__global__ __launch_bounds__(256)
void conv1x1_qkv_kernel(const _Float16* __restrict__ xm_t,
                        const _Float16* __restrict__ pq, const float* __restrict__ qb,
                        const _Float16* __restrict__ pk, const float* __restrict__ kb,
                        const _Float16* __restrict__ pv, const float* __restrict__ vb,
                        float* __restrict__ q, float* __restrict__ k, float* __restrict__ v)
{
    __shared__ _Float16 lb[64 * CP];    // 17408 B

    const int blk   = blockIdx.x;       // B*HW/64
    const size_t p0 = (size_t)blk * 64;
    const int b     = (int)(p0 / HW);
    const int phw   = (int)(p0 - (size_t)b * HW);
    const int tid   = threadIdx.x;
    const _Float16* src = xm_t + p0 * C_CH;

#if __has_builtin(__builtin_amdgcn_tensor_load_to_lds)
    if (tid < 32)
        tdm_load_2d(lb, src, /*tdim0=*/C_CH, /*tdim1=*/HW,
                    /*til0=*/C_CH, /*til1=*/64, /*stride0=*/C_CH);
#else
    for (int i = tid; i < 64 * 16; i += 256) {
        int chunk = i & 15, row = i >> 4;
        *(uint4*)&lb[(size_t)row * CP + chunk * 8] =
            *(const uint4*)&src[(size_t)row * C_CH + chunk * 8];
    }
#endif
    __syncthreads();

    const int lane = tid & 31;
    const int wave = tid >> 5;
    const int oc0  = wave * 16;
    const int hi   = lane >> 4;
    const int n    = lane & 15;

    v8f aq[4] = {}, ak[4] = {}, av[4] = {};
    const v16h* pAq = (const v16h*)pq;
    const v16h* pAk = (const v16h*)pk;
    const v16h* pAv = (const v16h*)pv;

    for (int cg = 0; cg < 4; ++cg) {
        const size_t fi = (size_t)((wave * 4 + cg) * 32 + lane);
        v16h fq = pAq[fi], fk = pAk[fi], fv = pAv[fi];
        const _Float16* lb0 = lb + (size_t)n * CP + cg * 32 + hi * 16;
#pragma unroll
        for (int nt = 0; nt < 4; ++nt) {
            v16h bf = lds_frag(lb0 + (size_t)nt * 16 * CP);
            aq[nt] = __builtin_amdgcn_wmma_f32_16x16x32_f16(false, fq, false, bf, (short)0, aq[nt], false, false);
            ak[nt] = __builtin_amdgcn_wmma_f32_16x16x32_f16(false, fk, false, bf, (short)0, ak[nt], false, false);
            av[nt] = __builtin_amdgcn_wmma_f32_16x16x32_f16(false, fv, false, bf, (short)0, av[nt], false, false);
        }
    }

    const size_t outbase = (size_t)b * CHW + phw;
#pragma unroll
    for (int vv = 0; vv < 8; ++vv) {
        const int M = vv + hi * 8;
        const float bq = qb[oc0 + M], bk2 = kb[oc0 + M], bv2 = vb[oc0 + M];
#pragma unroll
        for (int nt = 0; nt < 4; ++nt) {
            size_t o = outbase + (size_t)(oc0 + M) * HW + nt * 16 + n;
            q[o] = aq[nt][vv] + bq;
            k[o] = ak[nt][vv] + bk2;
            v[o] = av[nt][vv] + bv2;
        }
    }
}

// ---------------------------------------------------------------------------
// InstanceNorm stats: one block per (b,c); biased variance over HW.
// ---------------------------------------------------------------------------
__global__ __launch_bounds__(256)
void in_stats_kernel(const float* __restrict__ y, float* __restrict__ mu,
                     float* __restrict__ rs)
{
    const int bc = blockIdx.x;
    const float* p = y + (size_t)bc * HW;
    float s = 0.f, s2 = 0.f;
    for (int i = threadIdx.x; i < HW; i += 256) {
        float v = p[i];
        s += v; s2 += v * v;
    }
    __shared__ float sh[256], sh2[256];
    sh[threadIdx.x] = s; sh2[threadIdx.x] = s2;
    __syncthreads();
    for (int off = 128; off > 0; off >>= 1) {
        if (threadIdx.x < off) {
            sh[threadIdx.x]  += sh[threadIdx.x + off];
            sh2[threadIdx.x] += sh2[threadIdx.x + off];
        }
        __syncthreads();
    }
    if (threadIdx.x == 0) {
        float mean = sh[0] * (1.f / HW);
        float var  = sh2[0] * (1.f / HW) - mean * mean;
        mu[bc] = mean;
        rs[bc] = rsqrtf(var + EPSF);
    }
}

// IN + LeakyReLU -> f16 pixel-major with halo ring (border = 0)
__global__ __launch_bounds__(256)
void in_lrelu_pad_kernel(const float* __restrict__ y1, const float* __restrict__ mu,
                         const float* __restrict__ rs, _Float16* __restrict__ xin_t)
{
    size_t i = (size_t)blockIdx.x * 256 + threadIdx.x;
    if (i >= (size_t)B_DIM * PHW * C_CH) return;
    int c = (int)(i & (C_CH - 1));
    size_t pp = i >> 7;
    int b  = (int)(pp / PHW);
    int r  = (int)(pp - (size_t)b * PHW);
    int ph = r / PH, pwd = r - ph * PH;
    int h = ph - 1, w = pwd - 1;
    float v = 0.f;
    if ((unsigned)h < (unsigned)H_DIM && (unsigned)w < (unsigned)W_DIM) {
        int bc = b * C_CH + c;
        float o = (y1[(size_t)b * CHW + (size_t)c * HW + h * W_DIM + w] - mu[bc]) * rs[bc];
        v = (o >= 0.f) ? o : SLOPE * o;
    }
    xin_t[i] = (_Float16)v;
}

// xm_t[b][p][c] = xin_t(padded)[b][p][c] * sigmoid-mask (f32 NCHW)
__global__ __launch_bounds__(256)
void make_xm_kernel(const _Float16* __restrict__ xin_t, const float* __restrict__ maskb,
                    _Float16* __restrict__ xm_t)
{
    size_t i = (size_t)blockIdx.x * 256 + threadIdx.x;
    if (i >= NTOT) return;
    int c = (int)(i & (C_CH - 1));
    size_t pp = i >> 7;
    int b = (int)(pp / HW);
    int p = (int)(pp - (size_t)b * HW);
    int h = p >> 7, w = p & (W_DIM - 1);
    float m  = maskb[(size_t)b * CHW + (size_t)c * HW + p];
    float xv = (float)xin_t[((size_t)b * PHW + (size_t)(h + 1) * PH + (w + 1)) * C_CH + c];
    xm_t[i] = (_Float16)(xv * m);
}

// ---------------------------------------------------------------------------
// 3x3-window attention + final combine; one wave32 per pixel, 4 ch/lane.
// OOB shifted k/v collapse to their biases. out = x + mask*out1 + (1-mask)*out2
// ---------------------------------------------------------------------------
__global__ __launch_bounds__(256)
void attn_final_kernel(const float* __restrict__ x,  const float* __restrict__ out1,
                       const float* __restrict__ mask, const float* __restrict__ q,
                       const float* __restrict__ k,  const float* __restrict__ v,
                       const float* __restrict__ kbias, const float* __restrict__ vbias,
                       float* __restrict__ out)
{
    const int wave = threadIdx.x >> 5;
    const int lane = threadIdx.x & 31;
    const size_t p = (size_t)blockIdx.x * 8 + wave;
    const int b   = (int)(p / HW);
    const int phw = (int)(p - (size_t)b * HW);
    const int h   = phw >> 7;
    const int wpx = phw & (W_DIM - 1);
    const int c0  = lane * 4;

    const size_t base = (size_t)b * CHW + (size_t)c0 * HW + phw;

    float qv[4];
#pragma unroll
    for (int cc = 0; cc < 4; ++cc) qv[cc] = q[base + (size_t)cc * HW];

    float sims[9];
    int t = 0;
    for (int di = -1; di <= 1; ++di) {
        for (int dj = -1; dj <= 1; ++dj, ++t) {
            const int hh = h + di, ww = wpx + dj;
            float s = 0.f;
            if ((unsigned)hh < (unsigned)H_DIM && (unsigned)ww < (unsigned)W_DIM) {
                const size_t kb = (size_t)b * CHW + (size_t)c0 * HW + hh * W_DIM + ww;
#pragma unroll
                for (int cc = 0; cc < 4; ++cc) s += qv[cc] * k[kb + (size_t)cc * HW];
            } else {
#pragma unroll
                for (int cc = 0; cc < 4; ++cc) s += qv[cc] * kbias[c0 + cc];
            }
#pragma unroll
            for (int off = 16; off > 0; off >>= 1) s += __shfl_xor(s, off, 32);
            sims[t] = s;
        }
    }

    float mx = sims[0];
#pragma unroll
    for (int i = 1; i < 9; ++i) mx = fmaxf(mx, sims[i]);
    float e[9], sum = 0.f;
#pragma unroll
    for (int i = 0; i < 9; ++i) { e[i] = __expf(sims[i] - mx); sum += e[i]; }
    const float inv = 1.f / sum;

#pragma unroll
    for (int cc = 0; cc < 4; ++cc) {
        float o2 = 0.f;
        int tt = 0;
        for (int di = -1; di <= 1; ++di) {
            for (int dj = -1; dj <= 1; ++dj, ++tt) {
                const int hh = h + di, ww = wpx + dj;
                float val;
                if ((unsigned)hh < (unsigned)H_DIM && (unsigned)ww < (unsigned)W_DIM)
                    val = v[(size_t)b * CHW + (size_t)(c0 + cc) * HW + hh * W_DIM + ww];
                else
                    val = vbias[c0 + cc];
                o2 += e[tt] * inv * val;
            }
        }
        const size_t o = base + (size_t)cc * HW;
        const float mk = mask[o];
        out[o] = x[o] + mk * out1[o] + (1.f - mk) * o2;
    }
}

// ---------------------------------------------------------------------------
extern "C" void kernel_launch(void* const* d_in, const int* in_sizes, int n_in,
                              void* d_out, int out_size, void* d_ws, size_t ws_size,
                              hipStream_t stream) {
    const float* x       = (const float*)d_in[0];
    const float* conv1_w = (const float*)d_in[1];
    const float* conv1_b = (const float*)d_in[2];
    const float* conv2_w = (const float*)d_in[3];
    const float* conv2_b = (const float*)d_in[4];
    const float* mask_w  = (const float*)d_in[5];
    const float* mask_b  = (const float*)d_in[6];
    const float* q_w     = (const float*)d_in[7];
    const float* q_b     = (const float*)d_in[8];
    const float* k_w     = (const float*)d_in[9];
    const float* k_b     = (const float*)d_in[10];
    const float* v_w     = (const float*)d_in[11];
    const float* v_b     = (const float*)d_in[12];
    float* out = (float*)d_out;

    // ---- workspace layout ----
    float* y1    = (float*)d_ws;             // NTOT f32 (conv1 out)
    float* out1  = y1   + NTOT;
    float* maskb = out1 + NTOT;
    float* qv    = maskb + NTOT;
    float* kv    = qv   + NTOT;
    float* vv    = kv   + NTOT;
    float* mu    = vv   + NTOT;              // B*C
    float* rs    = mu + B_DIM * C_CH;
    _Float16* xin_t = (_Float16*)(rs + B_DIM * C_CH);   // B*PHW*C
    _Float16* xm_t  = xin_t + (size_t)B_DIM * PHW * C_CH;   // NTOT
    _Float16* pw1   = xm_t + NTOT;           // 147456 each (3x3)
    _Float16* pw2   = pw1 + 147456;
    _Float16* pwm   = pw2 + 147456;
    _Float16* pq    = pwm + 147456;          // 16384 each (1x1)
    _Float16* pk    = pq + 16384;
    _Float16* pv    = pk + 16384;

    const int convGrid = B_DIM * H_DIM * 2;              // 2048
    const int ewGrid   = (int)(NTOT / 256);              // 65536
    const int padGrid  = (int)(((size_t)B_DIM * PHW * C_CH + 255) / 256);

    // 0) pack all weights to f16 fragment layout
    pack_w_kernel<<<(147456 + 255) / 256, 256, 0, stream>>>(conv1_w, pw1, 9);
    pack_w_kernel<<<(147456 + 255) / 256, 256, 0, stream>>>(conv2_w, pw2, 9);
    pack_w_kernel<<<(147456 + 255) / 256, 256, 0, stream>>>(mask_w,  pwm, 9);
    pack_w_kernel<<<(16384 + 255) / 256, 256, 0, stream>>>(q_w, pq, 1);
    pack_w_kernel<<<(16384 + 255) / 256, 256, 0, stream>>>(k_w, pk, 1);
    pack_w_kernel<<<(16384 + 255) / 256, 256, 0, stream>>>(v_w, pv, 1);

    // 1) conv1 (f32 NCHW input path)
    conv3x3_wmma_kernel<0><<<convGrid, 256, 0, stream>>>(x, pw1, conv1_b, y1, 0);
    // 2) instance norm + leaky relu -> padded f16 pixel-major
    in_stats_kernel<<<B_DIM * C_CH, 256, 0, stream>>>(y1, mu, rs);
    in_lrelu_pad_kernel<<<padGrid, 256, 0, stream>>>(y1, mu, rs, xin_t);
    // 3) conv2 / mask (TDM-staged f16 path; sigmoid fused for mask)
    conv3x3_wmma_kernel<1><<<convGrid, 256, 0, stream>>>(xin_t, pw2, conv2_b, out1, 0);
    conv3x3_wmma_kernel<1><<<convGrid, 256, 0, stream>>>(xin_t, pwm, mask_b, maskb, 1);
    // 4) xm = xin * mask (pixel-major f16)
    make_xm_kernel<<<ewGrid, 256, 0, stream>>>(xin_t, maskb, xm_t);
    // 5) q,k,v fused 1x1 convs (TDM-staged)
    conv1x1_qkv_kernel<<<B_DIM * HW / 64, 256, 0, stream>>>(
        xm_t, pq, q_b, pk, k_b, pv, v_b, qv, kv, vv);
    // 6) local window attention + final combine + residual
    attn_final_kernel<<<B_DIM * HW / 8, 256, 0, stream>>>(
        x, out1, maskb, qv, kv, vv, k_b, v_b, out);
}
